// VectorQuantizer_62354335203565
// MI455X (gfx1250) — matrix operand order repaired and checked by reading, not compiled, not run on hardware
//
#include <hip/hip_runtime.h>
#include <hip/hip_bf16.h>

typedef __attribute__((ext_vector_type(16))) _Float16 v16h;
typedef __attribute__((ext_vector_type(8)))  _Float16 h8;
typedef __attribute__((ext_vector_type(4)))  _Float16 h4;
typedef __attribute__((ext_vector_type(8)))  float    v8f;
typedef int vq_v4i __attribute__((vector_size(16)));   // matches builtin param type

#define N_ROWS   16384      // 16*1024 rows of z
#define KDIM     256        // embedding dim
#define CODES    8192       // codebook size
#define Z_ELEMS  (N_ROWS * KDIM)      // 4,194,304
#define LOSS_SCALE (1.25f / 4194304.0f)   // (1+BETA)/numel
#define LDS_STRIDE 264      // 256 + 8 halves pad -> conflict-free ds_load_b128

// CDNA5 async global->LDS path (ASYNCcnt) if the toolchain declares it.
#if defined(__AMDGCN__) && __has_builtin(__builtin_amdgcn_global_load_async_to_lds_b128) && __has_builtin(__builtin_amdgcn_s_wait_asynccnt)
#define VQ_ASYNC_LDS 1
#else
#define VQ_ASYNC_LDS 0
#endif

union AFrag { v16h v; h8 h[2]; };

// ---------------------------------------------------------------- z -> f16
__global__ __launch_bounds__(256) void vq_cvt_z(const float* __restrict__ z,
                                                _Float16* __restrict__ zh) {
    int i = blockIdx.x * blockDim.x + threadIdx.x;   // one float4 per thread
    float4 v = ((const float4*)z)[i];
    h4 h = {(_Float16)v.x, (_Float16)v.y, (_Float16)v.z, (_Float16)v.w};
    ((h4*)zh)[i] = h;
}

// ------------------------------------------- embedding -> f16 + ||e||^2
__global__ __launch_bounds__(256) void vq_cvt_e(const float* __restrict__ e,
                                                _Float16* __restrict__ eh,
                                                float* __restrict__ enorm) {
    int w = threadIdx.x >> 5, l = threadIdx.x & 31;
    int row = blockIdx.x * 8 + w;                    // one wave per code row
    const float4* er = (const float4*)(e + (size_t)row * KDIM);
    h4* hr = (h4*)(eh + (size_t)row * KDIM);
    float acc = 0.f;
    #pragma unroll
    for (int t = l; t < 64; t += 32) {
        float4 v = er[t];
        acc += v.x*v.x + v.y*v.y + v.z*v.z + v.w*v.w;
        h4 h = {(_Float16)v.x, (_Float16)v.y, (_Float16)v.z, (_Float16)v.w};
        hr[t] = h;
    }
    #pragma unroll
    for (int off = 16; off; off >>= 1) acc += __shfl_xor(acc, off, 32);
    if (l == 0) enorm[row] = acc;
}

// ---------------------------------------------------------------- zero loss
__global__ void vq_zero_loss(float* out) { out[Z_ELEMS] = 0.0f; }

// ----------------------------------------- WMMA distance GEMM + row argmin
// block = 256 threads = 8 waves covering 128 rows; B code-tiles are staged
// once per workgroup in double-buffered LDS and shared by all 8 waves.
__global__ __launch_bounds__(256) void vq_argmin_wmma(const _Float16* __restrict__ zh,
                                                      const _Float16* __restrict__ eh,
                                                      const float* __restrict__ enorm,
                                                      int* __restrict__ idxout) {
    __shared__ __align__(16) _Float16 Bs[2][16 * LDS_STRIDE];   // 2 x ~8.25 KB

    const int tid  = threadIdx.x;
    const int w    = tid >> 5;
    const int l    = tid & 31;
    const int m    = l & 15;        // row-within-tile (A), col-within-tile (B/C)
    const int half = l >> 4;        // K-half selector (this code's contiguous k order)
    const int rowA = blockIdx.x * 128 + w * 16;

    // --- A fragments: 16x32 f16 tiles, all 8 K-blocks held in registers.
    // Lanes 0-15 hold K [kb*32+0..15]; lanes 16-31 hold
    // K [kb*32+16..31]; m = lane&15 is the row.
    AFrag a[8];
    const h8* zrow = (const h8*)(zh + (size_t)(rowA + m) * KDIM);
    #pragma unroll
    for (int kb = 0; kb < 8; ++kb) {
        a[kb].h[0] = zrow[kb * 4 + 2 * half];
        a[kb].h[1] = zrow[kb * 4 + 2 * half + 1];
    }

    // Stage one 16-code tile (8 KB) into LDS buffer `buf`: 512 x 16B chunks,
    // 2 chunks per thread, rows padded to LDS_STRIDE halves.
    auto stage = [&](int tile, int buf) {
        for (int c = tid; c < 512; c += 256) {
            const int row = c >> 5, off = c & 31;
            const _Float16* g = eh + (size_t)(tile * 16 + row) * KDIM + off * 8;
            _Float16* dst = &Bs[buf][row * LDS_STRIDE + off * 8];
#if VQ_ASYNC_LDS
            __builtin_amdgcn_global_load_async_to_lds_b128(
                (__attribute__((address_space(1))) vq_v4i*)g,
                (__attribute__((address_space(3))) vq_v4i*)dst, 0, 0);
#else
            *(uint4*)dst = *(const uint4*)g;     // global_load_b128 + ds_store_b128
#endif
        }
    };

    float best[8];
    int   bidx[8];
    #pragma unroll
    for (int r = 0; r < 8; ++r) { best[r] = 3.4e38f; bidx[r] = 0; }

    stage(0, 0);
#if VQ_ASYNC_LDS
    __builtin_amdgcn_s_wait_asynccnt(0);
#endif
    __syncthreads();

    for (int tn = 0; tn < CODES / 16; ++tn) {
        const int cur = tn & 1;
        if (tn + 1 < CODES / 16) stage(tn + 1, cur ^ 1);   // overlap with WMMAs

        // B fragment from LDS: lane = column (code) m, lane-half selects K 0-15/16-31;
        // per lane: 16 contiguous f16 at Bs[cur][m*STRIDE + kb*32 + half*16].
        const _Float16* brow = &Bs[cur][m * LDS_STRIDE + half * 16];
        v16h b[8];
        #pragma unroll
        for (int kb = 0; kb < 8; ++kb) b[kb] = *(const v16h*)(brow + kb * 32);

        v8f c = {};
        #pragma unroll
        for (int kb = 0; kb < 8; ++kb)
            c = __builtin_amdgcn_wmma_f32_16x16x32_f16(
                    false, a[kb].v, false, b[kb], (short)0, c, false, false);

        const int code = tn * 16 + m;
        const float en = enorm[code];
        // C layout: lanes 0-15 VGPR r -> row r; lanes 16-31 VGPR r -> row r+8.
        #pragma unroll
        for (int r = 0; r < 8; ++r) {
            float s = en - 2.0f * c[r];
            if (s < best[r]) { best[r] = s; bidx[r] = code; }  // codes ascend per lane
        }

#if VQ_ASYNC_LDS
        __builtin_amdgcn_s_wait_asynccnt(0);
#endif
        __syncthreads();
    }

    // Cross-lane argmin within each 16-lane half (same 8 rows per half),
    // first-index tie-break to match jnp.argmin.
    #pragma unroll
    for (int r = 0; r < 8; ++r) {
        float mv = best[r]; int mi = bidx[r];
        #pragma unroll
        for (int off = 8; off; off >>= 1) {
            float om = __shfl_xor(mv, off, 32);
            int   oi = __shfl_xor(mi, off, 32);
            if (om < mv || (om == mv && oi < mi)) { mv = om; mi = oi; }
        }
        bidx[r] = mi;
    }
    if (m == 0) {
        const int base = rowA + half * 8;
        #pragma unroll
        for (int r = 0; r < 8; ++r) idxout[base + r] = bidx[r];
    }
}

// ------------------------------------ gather z_q, loss, indices-as-float
__global__ __launch_bounds__(256) void vq_gather_loss(const float* __restrict__ z,
                                                      const float* __restrict__ emb,
                                                      const int* __restrict__ idx,
                                                      float* __restrict__ out) {
    int w = threadIdx.x >> 5, l = threadIdx.x & 31;
    int row = blockIdx.x * 8 + w;                    // one wave per z row
    int code = idx[row];
    const float4* er = (const float4*)(emb + (size_t)code * KDIM);
    const float4* zr = (const float4*)(z + (size_t)row * KDIM);
    float4* orow = (float4*)(out + (size_t)row * KDIM);
    float acc = 0.f;
    #pragma unroll
    for (int t = l; t < 64; t += 32) {
        float4 q = er[t], zz = zr[t];
        orow[t] = q;
        float dx = q.x - zz.x, dy = q.y - zz.y, dz = q.z - zz.z, dw = q.w - zz.w;
        acc += dx*dx + dy*dy + dz*dz + dw*dw;
    }
    #pragma unroll
    for (int off = 16; off; off >>= 1) acc += __shfl_xor(acc, off, 32);
    if (l == 0) {
        atomicAdd(out + Z_ELEMS, acc * LOSS_SCALE);  // global_atomic_add_f32
        out[Z_ELEMS + 1 + row] = (float)code;        // indices section
    }
}

extern "C" void kernel_launch(void* const* d_in, const int* in_sizes, int n_in,
                              void* d_out, int out_size, void* d_ws, size_t ws_size,
                              hipStream_t stream) {
    const float* z   = (const float*)d_in[0];   // (16,1024,256) f32
    const float* emb = (const float*)d_in[1];   // (8192,256) f32
    float* out = (float*)d_out;                 // [z_q | loss | indices]

    char* ws = (char*)d_ws;
    const size_t ZH_BYTES = (size_t)N_ROWS * KDIM * 2;   // 8 MB
    const size_t EH_BYTES = (size_t)CODES * KDIM * 2;    // 4 MB
    _Float16* zh    = (_Float16*)ws;
    _Float16* eh    = (_Float16*)(ws + ZH_BYTES);
    float*    enorm = (float*)(ws + ZH_BYTES + EH_BYTES);
    int*      idx   = (int*)(ws + ZH_BYTES + EH_BYTES + (size_t)CODES * 4);

    vq_cvt_z<<<Z_ELEMS / (256 * 4), 256, 0, stream>>>(z, zh);
    vq_cvt_e<<<CODES / 8, 256, 0, stream>>>(emb, eh, enorm);
    vq_zero_loss<<<1, 1, 0, stream>>>(out);
    vq_argmin_wmma<<<N_ROWS / 128, 256, 0, stream>>>(zh, eh, enorm, idx);
    vq_gather_loss<<<N_ROWS / 8, 256, 0, stream>>>(z, emb, idx, out);
}
